// Block_4492535792182
// MI455X (gfx1250) — matrix operand order repaired and checked by reading, not compiled, not run on hardware
//
#include <hip/hip_runtime.h>
#include <math.h>

#define DEV __device__ __forceinline__

constexpr int NB = 4, NT = 1024, NC = 1024, NH = 16, NN = 64;
constexpr int NROWS = NB * NT;   // 4096 token rows
constexpr int NFF   = 4 * NC;    // 4096 FFN dim
constexpr int DW_ = 64, DA_ = 64, DV_ = 64, DG_ = 160;

typedef __attribute__((ext_vector_type(16))) __bf16          v16bf;
typedef __attribute__((ext_vector_type(8)))  float           v8f;
typedef __attribute__((ext_vector_type(8)))  unsigned short  u16x8;

DEV unsigned short f2bf(float f) {
  unsigned int u = __builtin_bit_cast(unsigned int, f);
  u += 0x7fffu + ((u >> 16) & 1u);
  return (unsigned short)(u >> 16);
}

DEV float redu16(float v) {  // sum over aligned groups of 16 lanes
  v += __shfl_xor(v, 1); v += __shfl_xor(v, 2);
  v += __shfl_xor(v, 4); v += __shfl_xor(v, 8);
  return v;
}
DEV float redu4(float v) {   // sum over aligned groups of 4 lanes
  v += __shfl_xor(v, 1); v += __shfl_xor(v, 2);
  return v;
}

// ---------------------------------------------------------------- LayerNorm
__global__ __launch_bounds__(256) void ln_kernel(const float* __restrict__ x,
                                                 const float* __restrict__ g,
                                                 const float* __restrict__ b,
                                                 float* __restrict__ out, float eps) {
  int row = blockIdx.x, tid = threadIdx.x;
  size_t base = (size_t)row * NC;
  float v[4]; float s = 0.f, ss = 0.f;
#pragma unroll
  for (int k = 0; k < 4; ++k) { v[k] = x[base + k * 256 + tid]; s += v[k]; ss += v[k] * v[k]; }
#pragma unroll
  for (int o = 1; o < 32; o <<= 1) { s += __shfl_xor(s, o); ss += __shfl_xor(ss, o); }
  __shared__ float ls[8], lss[8];
  int wave = tid >> 5, lane = tid & 31;
  if (lane == 0) { ls[wave] = s; lss[wave] = ss; }
  __syncthreads();
  s = 0.f; ss = 0.f;
#pragma unroll
  for (int w = 0; w < 8; ++w) { s += ls[w]; ss += lss[w]; }
  float mean = s * (1.0f / NC);
  float var  = ss * (1.0f / NC) - mean * mean;
  float inv  = rsqrtf(var + eps);
#pragma unroll
  for (int k = 0; k < 4; ++k) {
    int c = k * 256 + tid;
    out[base + c] = (v[k] - mean) * inv * g[c] + b[c];
  }
}

// ------------------------------------------------ token-shift mixes (6-way)
__global__ __launch_bounds__(256) void mix6_kernel(const float* __restrict__ h,
    const float* __restrict__ cr, const float* __restrict__ cw, const float* __restrict__ ck,
    const float* __restrict__ cv, const float* __restrict__ ca, const float* __restrict__ cg,
    unsigned short* __restrict__ oR, unsigned short* __restrict__ oW, unsigned short* __restrict__ oK,
    unsigned short* __restrict__ oV, unsigned short* __restrict__ oA, unsigned short* __restrict__ oG) {
  int row = blockIdx.x; int t = row & (NT - 1);
  size_t base = (size_t)row * NC;
#pragma unroll
  for (int k = 0; k < 4; ++k) {
    int c = k * 256 + threadIdx.x;
    float ht = h[base + c];
    float hp = (t == 0) ? 0.f : h[base - NC + c];
    float xx = hp - ht;
    oR[base + c] = f2bf(ht + xx * cr[c]);
    oW[base + c] = f2bf(ht + xx * cw[c]);
    oK[base + c] = f2bf(ht + xx * ck[c]);
    oV[base + c] = f2bf(ht + xx * cv[c]);
    oA[base + c] = f2bf(ht + xx * ca[c]);
    oG[base + c] = f2bf(ht + xx * cg[c]);
  }
}

__global__ __launch_bounds__(256) void mix1_kernel(const float* __restrict__ h,
    const float* __restrict__ cc, unsigned short* __restrict__ o) {
  int row = blockIdx.x; int t = row & (NT - 1);
  size_t base = (size_t)row * NC;
#pragma unroll
  for (int k = 0; k < 4; ++k) {
    int c = k * 256 + threadIdx.x;
    float ht = h[base + c];
    float hp = (t == 0) ? 0.f : h[base - NC + c];
    o[base + c] = f2bf(ht + (hp - ht) * cc[c]);
  }
}

// -------------------------------------------------- weight convert/transpose
__global__ void f32_to_bf16_kernel(const float* __restrict__ in, unsigned short* __restrict__ out, size_t n) {
  size_t i = (size_t)blockIdx.x * blockDim.x + threadIdx.x;
  if (i < n) out[i] = f2bf(in[i]);
}
__global__ void transpose_bf16_kernel(const float* __restrict__ in,  // [K,N] f32
                                      unsigned short* __restrict__ out, int K, int N) {  // [N,K] bf16
  size_t i = (size_t)blockIdx.x * blockDim.x + threadIdx.x;
  if (i < (size_t)K * N) {
    int k = (int)(i / N), n = (int)(i % N);
    out[(size_t)n * K + k] = f2bf(in[i]);
  }
}

// ------------------------------------------------------------- WMMA GEMM
// out[M,N] = act(A[M,K] * Bt[N,K]^T + bias) (+ resid). A,Bt bf16, acc f32.
// Block tile 128x64, 8 waves, wave tile 32x32 (4 WMMA / K-step, B-frags reused),
// LDS double-buffered: one barrier per K-step, global loads overlapped with WMMA.
enum { ACT_NONE = 0, ACT_TANH = 1, ACT_SIGMOID = 2, ACT_DECAY = 3, ACT_RELUSQ = 4 };

template <int ACT> DEV float apply_act(float v) {
  if (ACT == ACT_TANH)    v = tanhf(v);
  if (ACT == ACT_SIGMOID) v = 1.0f / (1.0f + expf(-v));
  if (ACT == ACT_DECAY)   { float w = -log1pf(expf(-v)) - 0.5f; v = expf(-expf(w)); }
  if (ACT == ACT_RELUSQ)  { float m = fmaxf(v, 0.f); v = m * m; }
  return v;
}

template <int ACT, bool OBF16, bool RESID, bool BIAS>
__global__ __launch_bounds__(256) void gemm_bf16_wmma(
    const unsigned short* __restrict__ A,   // [M,K] bf16 bits
    const unsigned short* __restrict__ Bt,  // [N,K] bf16 bits
    void* __restrict__ Out,                 // [M,N]
    const float* __restrict__ bias,         // [N] or null
    const float* __restrict__ resid,        // [M,N] or null
    int M, int N, int K) {
  __shared__ __align__(16) unsigned short As[2][128][40];
  __shared__ __align__(16) unsigned short Bs[2][64][40];

  const int tid = threadIdx.x, lane = tid & 31, wave = tid >> 5;
  const int m0 = blockIdx.y * 128, n0 = blockIdx.x * 64;
  const int wm = (wave >> 1) * 32;  // 0/32/64/96
  const int wn = (wave & 1) * 32;   // 0/32
  const int lane16 = lane & 15, laneh = lane >> 4;

  v8f accs[2][2] = {};

  // staging regs for the global->LDS pipeline
  uint4 ra0, ra1, rb0;
  const int sRow = tid >> 2;          // 0..63
  const int sSeg = (tid & 3) * 8;     // halves: 0,8,16,24

  auto load_regs = [&](int kk) {
    ra0 = *(const uint4*)(A + (size_t)(m0 + sRow) * K + kk + sSeg);
    ra1 = *(const uint4*)(A + (size_t)(m0 + 64 + sRow) * K + kk + sSeg);
    rb0.x = rb0.y = rb0.z = rb0.w = 0u;
    int n = n0 + sRow;
    if (n < N) rb0 = *(const uint4*)(Bt + (size_t)n * K + kk + sSeg);
  };
  auto store_lds = [&](int buf) {
    *(uint4*)&As[buf][sRow][sSeg]      = ra0;
    *(uint4*)&As[buf][64 + sRow][sSeg] = ra1;
    *(uint4*)&Bs[buf][sRow][sSeg]      = rb0;
  };
  // A: lane = row (M), K split 8+8 at +16 (ISA 7.12.2, 16-bit A layout)
  auto load_afrag = [&](int buf, int r) {
    int kb = laneh * 8;
    u16x8 lo = *(const u16x8*)&As[buf][r + lane16][kb];
    u16x8 hi = *(const u16x8*)&As[buf][r + lane16][kb + 16];
    v16bf f;
#pragma unroll
    for (int i = 0; i < 8; ++i) {
      f[i]     = __builtin_bit_cast(__bf16, (unsigned short)lo[i]);
      f[i + 8] = __builtin_bit_cast(__bf16, (unsigned short)hi[i]);
    }
    return f;
  };
  // B: lane = column (N), K split 8+8 at +16 (same order as A)
  auto load_bfrag = [&](int buf, int c) {
    int kb = laneh * 8;
    u16x8 lo = *(const u16x8*)&Bs[buf][c + lane16][kb];
    u16x8 hi = *(const u16x8*)&Bs[buf][c + lane16][kb + 16];
    v16bf f;
#pragma unroll
    for (int i = 0; i < 8; ++i) {
      f[i]     = __builtin_bit_cast(__bf16, (unsigned short)lo[i]);
      f[i + 8] = __builtin_bit_cast(__bf16, (unsigned short)hi[i]);
    }
    return f;
  };

  load_regs(0);
  store_lds(0);
  __syncthreads();

  int cur = 0;
  for (int kk = 0; kk < K; kk += 32) {
    const bool more = (kk + 32) < K;
    if (more) load_regs(kk + 32);  // overlap global latency with WMMA below

    v16bf b0 = load_bfrag(cur, wn);
    v16bf b1 = load_bfrag(cur, wn + 16);
#pragma unroll
    for (int ai = 0; ai < 2; ++ai) {
      v16bf af = load_afrag(cur, wm + 16 * ai);
      accs[ai][0] = __builtin_amdgcn_wmma_f32_16x16x32_bf16(false, af, false, b0,
                                                            (short)0, accs[ai][0], false, false);
      accs[ai][1] = __builtin_amdgcn_wmma_f32_16x16x32_bf16(false, af, false, b1,
                                                            (short)0, accs[ai][1], false, false);
    }
    if (more) store_lds(cur ^ 1);
    __syncthreads();
    cur ^= 1;
  }

  // C/D: VGPR g -> M = g + 8*(lane>=16), N = lane%16
#pragma unroll
  for (int ai = 0; ai < 2; ++ai) {
    const int rbase = m0 + wm + 16 * ai + laneh * 8;
#pragma unroll
    for (int bi = 0; bi < 2; ++bi) {
      int col = n0 + wn + 16 * bi + lane16;
      if (col < N) {
        float bv = BIAS ? bias[col] : 0.f;
#pragma unroll
        for (int gI = 0; gI < 8; ++gI) {
          int row = rbase + gI;
          float val = apply_act<ACT>(accs[ai][bi][gI] + bv);
          if (RESID) val += resid[(size_t)row * N + col];
          if (OBF16) ((unsigned short*)Out)[(size_t)row * N + col] = f2bf(val);
          else       ((float*)Out)[(size_t)row * N + col] = val;
        }
      }
    }
  }
}

// ------------------------------------------------------- WKV prep (per row)
__global__ __launch_bounds__(256) void prep_wkv_kernel(
    const float* __restrict__ r, float* __restrict__ k, float* __restrict__ v,
    const float* __restrict__ dec, const float* __restrict__ a, const float* __restrict__ vmix,
    const float* __restrict__ vfirst, const float* __restrict__ k_k, const float* __restrict__ k_a,
    float* __restrict__ rT, float* __restrict__ dT, float* __restrict__ kT,
    float* __restrict__ vT, float* __restrict__ aT, float* __restrict__ bT) {
  int row = blockIdx.x, b = row >> 10, t = row & (NT - 1);
  int c0 = threadIdx.x * 4, h = c0 >> 6;
  size_t gi = (size_t)row * NC + c0;
  float kv[4], av[4], kkv[4];
  float nrm = 0.f;
#pragma unroll
  for (int j = 0; j < 4; ++j) {
    kv[j]  = k[gi + j];
    av[j]  = a[gi + j];
    kkv[j] = kv[j] * k_k[c0 + j];
    nrm += kkv[j] * kkv[j];
  }
  nrm = redu16(nrm);
  float inv = 1.0f / fmaxf(sqrtf(nrm), 1e-12f);
  size_t si = (((size_t)(b * NH + h)) * NT + t) * NN + (c0 & 63);
#pragma unroll
  for (int j = 0; j < 4; ++j) {
    float kkn = kkv[j] * inv;
    float k2  = kv[j] * (1.f + (av[j] - 1.f) * k_a[c0 + j]);
    float vm  = vmix[gi + j];
    float vv  = v[gi + j];
    float vmx = vv + (vfirst[gi + j] - vv) * vm;
    k[gi + j] = k2;            // in-place: becomes k' for bonus term
    v[gi + j] = vmx;           // in-place: mixed v
    rT[si + j] = r[gi + j];
    dT[si + j] = dec[gi + j];
    kT[si + j] = k2;
    vT[si + j] = vmx;
    aT[si + j] = -kkn;
    bT[si + j] = kkn * av[j];
  }
}

// -------------------------------------------------- WKV7 sequential scan
// One block per (b,h). State S[i][j] 64x64 f32 in registers:
// thread (i = tid/4) holds j-slice [16*(tid&3), +16).
// Double-buffered LDS staging: step t+1 prefetched (float4, 96 lanes) while
// computing step t; single barrier per step.
__global__ __launch_bounds__(256) void wkv7_scan_kernel(
    const float* __restrict__ rT, const float* __restrict__ dT,
    const float* __restrict__ kT, const float* __restrict__ vT,
    const float* __restrict__ aT, const float* __restrict__ bT,
    float* __restrict__ y) {  // [B,T,C]
  const int bh = blockIdx.x;
  const int b = bh >> 4, h = bh & 15;
  const int tid = threadIdx.x;
  const int i = tid >> 2, q = tid & 3, jb = q * 16;

  __shared__ __align__(16) float sb[2][6][64];  // [buf][r,d,k,v,a,b][64]

  float S[16];
#pragma unroll
  for (int j = 0; j < 16; ++j) S[j] = 0.f;

  const size_t base0 = (size_t)bh * NT * NN;
  const int arr = tid >> 4;          // 0..5 for tid<96
  const int e4  = (tid & 15) * 4;

  auto load_t = [&](int t, int buf) {
    if (tid < 96) {
      const float* src = (arr == 0) ? rT : (arr == 1) ? dT : (arr == 2) ? kT
                        : (arr == 3) ? vT : (arr == 4) ? aT : bT;
      *(float4*)&sb[buf][arr][e4] = *(const float4*)(src + base0 + (size_t)t * NN + e4);
    }
  };

  load_t(0, 0);
  __syncthreads();

  for (int t = 0; t < NT; ++t) {
    const int cur = t & 1;
    if (t + 1 < NT) load_t(t + 1, cur ^ 1);

    float sap = 0.f;
#pragma unroll
    for (int j = 0; j < 16; ++j) sap += S[j] * sb[cur][4][jb + j];
    sap = redu4(sap);
    float vi = sb[cur][3][i];
    float outp = 0.f;
#pragma unroll
    for (int j = 0; j < 16; ++j) {
      float s = S[j] * sb[cur][1][jb + j] + sap * sb[cur][5][jb + j] + vi * sb[cur][2][jb + j];
      S[j] = s;
      outp += s * sb[cur][0][jb + j];
    }
    outp = redu4(outp);
    if (q == 0) y[((size_t)(b * NT + t)) * NC + h * NN + i] = outp;
    __syncthreads();
  }
}

// --------------------------------- post: GroupNorm + bonus + *g -> bf16
__global__ __launch_bounds__(256) void post_kernel(
    const float* __restrict__ y, const float* __restrict__ g, const float* __restrict__ r,
    const float* __restrict__ k2, const float* __restrict__ v, const float* __restrict__ r_k,
    const float* __restrict__ lnx_g, const float* __restrict__ lnx_b,
    unsigned short* __restrict__ yg_bf) {
  int row = blockIdx.x;
  int c0 = threadIdx.x * 4, h = c0 >> 6, n = c0 & 63;
  size_t gi = (size_t)row * NC + c0;
  float yv[4];
  float s = 0.f, ss = 0.f, bp = 0.f;
#pragma unroll
  for (int j = 0; j < 4; ++j) {
    yv[j] = y[gi + j];
    s += yv[j]; ss += yv[j] * yv[j];
    bp += r[gi + j] * k2[gi + j] * r_k[h * NN + n + j];
  }
  s = redu16(s); ss = redu16(ss); bp = redu16(bp);
  float mean = s * (1.0f / NN);
  float var  = ss * (1.0f / NN) - mean * mean;
  float inv  = rsqrtf(var + 0.00064f);
#pragma unroll
  for (int j = 0; j < 4; ++j) {
    int c = c0 + j;
    float val = (yv[j] - mean) * inv * lnx_g[c] + lnx_b[c] + bp * v[gi + j];
    yg_bf[gi + j] = f2bf(val * g[gi + j]);
  }
}

// ================================================================= host
extern "C" void kernel_launch(void* const* d_in, const int* in_sizes, int n_in,
                              void* d_out, int out_size, void* d_ws, size_t ws_size,
                              hipStream_t stream) {
  const float* x      = (const float*)d_in[0];
  const float* vfirst = (const float*)d_in[1];
  const float* ln1_g  = (const float*)d_in[2];
  const float* ln1_b  = (const float*)d_in[3];
  const float* ln2_g  = (const float*)d_in[4];
  const float* ln2_b  = (const float*)d_in[5];
  const float* x_r    = (const float*)d_in[6];
  const float* x_w    = (const float*)d_in[7];
  const float* x_kmix = (const float*)d_in[8];
  const float* x_vmix = (const float*)d_in[9];
  const float* x_a    = (const float*)d_in[10];
  const float* x_g    = (const float*)d_in[11];
  const float* w0 = (const float*)d_in[12];
  const float* w1 = (const float*)d_in[13];
  const float* w2 = (const float*)d_in[14];
  const float* a0 = (const float*)d_in[15];
  const float* a1 = (const float*)d_in[16];
  const float* a2 = (const float*)d_in[17];
  const float* v0 = (const float*)d_in[18];
  const float* v1 = (const float*)d_in[19];
  const float* v2 = (const float*)d_in[20];
  const float* g1 = (const float*)d_in[21];
  const float* g2 = (const float*)d_in[22];
  const float* k_k = (const float*)d_in[23];
  const float* k_a = (const float*)d_in[24];
  const float* r_k = (const float*)d_in[25];
  const float* W_r = (const float*)d_in[26];
  const float* W_k = (const float*)d_in[27];
  const float* W_v = (const float*)d_in[28];
  const float* W_o = (const float*)d_in[29];
  const float* lnx_g = (const float*)d_in[30];
  const float* lnx_b = (const float*)d_in[31];
  const float* xk_c  = (const float*)d_in[32];
  const float* Wk_ff = (const float*)d_in[33];
  const float* Wv_ff = (const float*)d_in[34];

  char* W = (char*)d_ws;
  size_t off = 0;
  auto take = [&](size_t bytes) -> char* {
    char* p = W + off;
    off += (bytes + 255) & ~(size_t)255;
    return p;
  };
  const size_t F = (size_t)NROWS * NC;  // 4,194,304 elements

  float* h            = (float*)take(F * 4);           // also reused as h2
  unsigned short* mR  = (unsigned short*)take(F * 2);
  unsigned short* mW  = (unsigned short*)take(F * 2);  // reused for CMix kf_in
  unsigned short* mK  = (unsigned short*)take(F * 2);
  unsigned short* mV  = (unsigned short*)take(F * 2);
  unsigned short* mA  = (unsigned short*)take(F * 2);
  unsigned short* mG  = (unsigned short*)take(F * 2);
  float* rbuf   = (float*)take(F * 4);
  float* kbuf   = (float*)take(F * 4);   // k -> k' in place
  float* vbuf   = (float*)take(F * 4);   // v -> v-mixed in place
  float* decbuf = (float*)take(F * 4);   // dec; reused as y after prep
  float* abuf   = (float*)take(F * 4);   // a;   reused as x1 after prep
  float* vmixb  = (float*)take(F * 4);   // vmix; low half reused as yg bf16
  float* gbuf   = (float*)take(F * 4);
  unsigned short* loraT = (unsigned short*)take((size_t)NROWS * DG_ * 2);
  unsigned short* w1T = (unsigned short*)take((size_t)DW_ * NC * 2);
  unsigned short* w2T = (unsigned short*)take((size_t)NC * DW_ * 2);
  unsigned short* a1T = (unsigned short*)take((size_t)DA_ * NC * 2);
  unsigned short* a2T = (unsigned short*)take((size_t)NC * DA_ * 2);
  unsigned short* v1T = (unsigned short*)take((size_t)DV_ * NC * 2);
  unsigned short* v2T = (unsigned short*)take((size_t)NC * DV_ * 2);
  unsigned short* g1T = (unsigned short*)take((size_t)DG_ * NC * 2);
  unsigned short* g2T = (unsigned short*)take((size_t)NC * DG_ * 2);
  unsigned short* WrB  = (unsigned short*)take((size_t)NC * NC * 2);
  unsigned short* WkB  = (unsigned short*)take((size_t)NC * NC * 2);
  unsigned short* WvB  = (unsigned short*)take((size_t)NC * NC * 2);
  unsigned short* WoB  = (unsigned short*)take((size_t)NC * NC * 2);
  unsigned short* WkfB = (unsigned short*)take((size_t)NFF * NC * 2);
  unsigned short* WvfB = (unsigned short*)take((size_t)NC * NFF * 2);
  float* rT = (float*)take(F * 4);
  float* dT = (float*)take(F * 4);
  float* kT = (float*)take(F * 4);
  float* vT = (float*)take(F * 4);
  float* aT = (float*)take(F * 4);
  float* bT = (float*)take(F * 4);
  unsigned short* kf1 = (unsigned short*)take((size_t)NROWS * NFF * 2);

  float* ybuf = decbuf;                           // reuse
  float* x1   = abuf;                             // reuse
  unsigned short* ygBF = (unsigned short*)vmixb;  // reuse (8MB of 16MB)
  unsigned short* kfin = mW;                      // reuse

  auto cvt = [&](const float* src, unsigned short* dst, size_t n) {
    f32_to_bf16_kernel<<<dim3((unsigned)((n + 255) / 256)), dim3(256), 0, stream>>>(src, dst, n);
  };
  auto tra = [&](const float* src, unsigned short* dst, int Kd, int Nd) {
    size_t n = (size_t)Kd * Nd;
    transpose_bf16_kernel<<<dim3((unsigned)((n + 255) / 256)), dim3(256), 0, stream>>>(src, dst, Kd, Nd);
  };

  // weights -> bf16 (W already [N,K] for the *.T matmuls)
  cvt(W_r, WrB, (size_t)NC * NC);
  cvt(W_k, WkB, (size_t)NC * NC);
  cvt(W_v, WvB, (size_t)NC * NC);
  cvt(W_o, WoB, (size_t)NC * NC);
  cvt(Wk_ff, WkfB, (size_t)NFF * NC);
  cvt(Wv_ff, WvfB, (size_t)NC * NFF);
  // LoRA weights [K,N] -> [N,K]
  tra(w1, w1T, NC, DW_);  tra(w2, w2T, DW_, NC);
  tra(a1, a1T, NC, DA_);  tra(a2, a2T, DA_, NC);
  tra(v1, v1T, NC, DV_);  tra(v2, v2T, DV_, NC);
  tra(g1, g1T, NC, DG_);  tra(g2, g2T, DG_, NC);

  // LN1 + mixes
  ln_kernel<<<NROWS, 256, 0, stream>>>(x, ln1_g, ln1_b, h, 1e-5f);
  mix6_kernel<<<NROWS, 256, 0, stream>>>(h, x_r, x_w, x_kmix, x_vmix, x_a, x_g,
                                         mR, mW, mK, mV, mA, mG);

  auto grid = [](int N) { return dim3((unsigned)((N + 63) / 64), NROWS / 128); };

  // projections
  gemm_bf16_wmma<ACT_NONE, false, false, false><<<grid(NC), 256, 0, stream>>>(
      mR, WrB, rbuf, nullptr, nullptr, NROWS, NC, NC);
  gemm_bf16_wmma<ACT_NONE, false, false, false><<<grid(NC), 256, 0, stream>>>(
      mK, WkB, kbuf, nullptr, nullptr, NROWS, NC, NC);
  gemm_bf16_wmma<ACT_NONE, false, false, false><<<grid(NC), 256, 0, stream>>>(
      mV, WvB, vbuf, nullptr, nullptr, NROWS, NC, NC);

  // w LoRA: dec = exp(-exp(-softplus(-(w0 + tanh(xw@w1)@w2)) - 0.5))
  gemm_bf16_wmma<ACT_TANH, true, false, false><<<grid(DW_), 256, 0, stream>>>(
      mW, w1T, loraT, nullptr, nullptr, NROWS, DW_, NC);
  gemm_bf16_wmma<ACT_DECAY, false, false, true><<<grid(NC), 256, 0, stream>>>(
      loraT, w2T, decbuf, w0, nullptr, NROWS, NC, DW_);
  // a LoRA: sigmoid(a0 + (xa@a1)@a2)
  gemm_bf16_wmma<ACT_NONE, true, false, false><<<grid(DA_), 256, 0, stream>>>(
      mA, a1T, loraT, nullptr, nullptr, NROWS, DA_, NC);
  gemm_bf16_wmma<ACT_SIGMOID, false, false, true><<<grid(NC), 256, 0, stream>>>(
      loraT, a2T, abuf, a0, nullptr, NROWS, NC, DA_);
  // v LoRA: sigmoid(v0 + (xv@v1)@v2)
  gemm_bf16_wmma<ACT_NONE, true, false, false><<<grid(DV_), 256, 0, stream>>>(
      mV, v1T, loraT, nullptr, nullptr, NROWS, DV_, NC);
  gemm_bf16_wmma<ACT_SIGMOID, false, false, true><<<grid(NC), 256, 0, stream>>>(
      loraT, v2T, vmixb, v0, nullptr, NROWS, NC, DV_);
  // g LoRA: sigmoid(xg@g1)@g2
  gemm_bf16_wmma<ACT_SIGMOID, true, false, false><<<grid(DG_), 256, 0, stream>>>(
      mG, g1T, loraT, nullptr, nullptr, NROWS, DG_, NC);
  gemm_bf16_wmma<ACT_NONE, false, false, false><<<grid(NC), 256, 0, stream>>>(
      loraT, g2T, gbuf, nullptr, nullptr, NROWS, NC, DG_);

  // WKV prep -> scan -> post
  prep_wkv_kernel<<<NROWS, 256, 0, stream>>>(rbuf, kbuf, vbuf, decbuf, abuf, vmixb,
                                             vfirst, k_k, k_a, rT, dT, kT, vT, aT, bT);
  wkv7_scan_kernel<<<NB * NH, 256, 0, stream>>>(rT, dT, kT, vT, aT, bT, ybuf);
  post_kernel<<<NROWS, 256, 0, stream>>>(ybuf, gbuf, rbuf, kbuf, vbuf, r_k,
                                         lnx_g, lnx_b, ygBF);

  // x1 = x + (y*g) @ W_o^T
  gemm_bf16_wmma<ACT_NONE, false, true, false><<<grid(NC), 256, 0, stream>>>(
      ygBF, WoB, x1, nullptr, x, NROWS, NC, NC);

  // CMix
  ln_kernel<<<NROWS, 256, 0, stream>>>(x1, ln2_g, ln2_b, h, 1e-5f);
  mix1_kernel<<<NROWS, 256, 0, stream>>>(h, xk_c, kfin);
  gemm_bf16_wmma<ACT_RELUSQ, true, false, false><<<grid(NFF), 256, 0, stream>>>(
      kfin, WkfB, kf1, nullptr, nullptr, NROWS, NFF, NC);
  gemm_bf16_wmma<ACT_NONE, false, true, false><<<grid(NC), 256, 0, stream>>>(
      kf1, WvfB, (float*)d_out, nullptr, x1, NROWS, NC, NFF);
}
